// Conv3d_24696061952249
// MI455X (gfx1250) — compile-verified, matrix operand order checked by reading, not run on hardware
//
#include <hip/hip_runtime.h>
#include <hip/hip_bf16.h>
#include <cstdint>

typedef __attribute__((ext_vector_type(16))) __bf16 v16bf;
typedef __attribute__((ext_vector_type(8)))  __bf16 v8bf;
typedef __attribute__((ext_vector_type(8)))  float  v8f;
typedef __attribute__((ext_vector_type(4)))  float  f4;

#define BATCH   2
#define CIN     128
#define COUT    256
#define FW      49          // filtered extent
#define OW      24          // output extent
#define SPATIAL (OW*OW*OW)  // 13824
#define TAPELEMS (COUT*CIN) // 32768 bf16 per tap = 64 KB

__device__ __forceinline__ __bf16 f2bf(float f) {
  uint32_t u = __builtin_bit_cast(uint32_t, f);
  u += 0x7FFFu + ((u >> 16) & 1u);              // round-to-nearest-even
  uint16_t h = (uint16_t)(u >> 16);
  return __builtin_bit_cast(__bf16, h);
}

// ---------------- Pass 1: FIR along W + transpose NCDHW -> (b,d,h,w,c) ----------------
__global__ __launch_bounds__(256) void fir_w_tr(const float* __restrict__ x,
                                                float* __restrict__ t1) {
  __shared__ float lds[32 * 49];                 // stride 49: conflict-free c-major reads
  int bid = blockIdx.x;
  int cg = bid & 3;  int rest = bid >> 2;
  int h  = rest % 48; rest /= 48;
  int d  = rest % 48;
  int b  = rest / 48;
  int tid = threadIdx.x;

  const float* xin = x + (((size_t)(b * CIN + cg * 32) * 48 + d) * 48 + h) * 48;
  #pragma unroll
  for (int j = 0; j < 6; ++j) {                  // 32 ch x 48 w = 1536 = 6*256
    int idx = j * 256 + tid;
    int cl = idx / 48, wi = idx % 48;
    lds[cl * 49 + wi] = xin[(size_t)cl * (48 * 48 * 48) + wi];
  }
  __syncthreads();

  const float KW[5] = {1.f/11.f, 3.f/11.f, 3.f/11.f, 3.f/11.f, 1.f/11.f};
  float* tout = t1 + (((size_t)(b * 48 + d) * 48 + h) * 49) * 128 + cg * 32;
  for (int j = 0; j < 7; ++j) {                  // 32 ch x 49 w = 1568
    int idx = j * 256 + tid;
    if (idx < 32 * 49) {
      int cl = idx & 31, wo = idx >> 5;
      float acc = 0.f;
      #pragma unroll
      for (int t = 0; t < 5; ++t) {
        int wi = wo + t - 3;
        if (wi >= 0 && wi < 48) acc += KW[t] * lds[cl * 49 + wi];
      }
      tout[(size_t)wo * 128 + cl] = acc;         // c-contiguous store
    }
  }
}

// ---------------- Pass 2: FIR along H, channels-last, fully coalesced ----------------
__global__ __launch_bounds__(256) void fir_h(const float* __restrict__ t1,
                                             float* __restrict__ t2) {
  size_t o = (size_t)blockIdx.x * 256 + threadIdx.x;     // (b,48,49,49,128)
  int c = (int)(o & 127); size_t r = o >> 7;
  int w  = (int)(r % 49); r /= 49;
  int ho = (int)(r % 49); r /= 49;
  int d  = (int)(r % 48);
  int b  = (int)(r / 48);
  const float KW[5] = {1.f/11.f, 3.f/11.f, 3.f/11.f, 3.f/11.f, 1.f/11.f};
  float acc = 0.f;
  #pragma unroll
  for (int t = 0; t < 5; ++t) {
    int hi = ho + t - 3;
    if (hi >= 0 && hi < 48)
      acc += KW[t] * t1[(((size_t)(b * 48 + d) * 48 + hi) * 49 + w) * 128 + c];
  }
  t2[o] = acc;
}

// ---------------- Pass 3: FIR along D + convert to bf16 channels-last ----------------
__global__ __launch_bounds__(256) void fir_d_bf16(const float* __restrict__ t2,
                                                  __bf16* __restrict__ xf) {
  size_t o = (size_t)blockIdx.x * 256 + threadIdx.x;     // (b,49,49,49,128)
  int c = (int)(o & 127); size_t r = o >> 7;
  int w  = (int)(r % 49); r /= 49;
  int h  = (int)(r % 49); r /= 49;
  int dd = (int)(r % 49);
  int b  = (int)(r / 49);
  const float KW[5] = {1.f/11.f, 3.f/11.f, 3.f/11.f, 3.f/11.f, 1.f/11.f};
  float acc = 0.f;
  #pragma unroll
  for (int t = 0; t < 5; ++t) {
    int di = dd + t - 3;
    if (di >= 0 && di < 48)
      acc += KW[t] * t2[(((size_t)(b * 48 + di) * 49 + h) * 49 + w) * 128 + c];
  }
  xf[o] = f2bf(acc);
}

// ---------------- Weight repack: (co,ci,3,3,3) f32 -> per-tap WMMA-fragment order ----
// Dest layout: [tap 27][c0g 4][colgroup 16][lane 32][elem 16] bf16.
// A lane's B-fragment (16 bf16 = 32 B) is contiguous, so the async copy to LDS is a
// straight linear memcpy and LDS reads are 2x ds_read_b128 at 32 B lane stride.
__global__ __launch_bounds__(256) void wprep(const float* __restrict__ w,
                                             __bf16* __restrict__ wt) {
  int o = blockIdx.x * 256 + threadIdx.x;
  int pos  = o & 15;
  int lane = (o >> 4) & 31;
  int cg   = (o >> 9) & 15;
  int c0g  = (o >> 13) & 3;
  int t    = o >> 15;
  int rrow = lane & 15;
  int hi   = lane >> 4;
  int n    = cg * 16 + rrow;                     // cout column
  // fragment elems 0..7 -> K = klo..klo+7 ; elems 8..15 -> K = klo+16..klo+23
  int cw   = (pos < 8) ? (hi * 8 + pos) : (16 + hi * 8 + (pos - 8));
  int c    = c0g * 32 + cw;                      // cin
  wt[o] = f2bf(w[((size_t)n * CIN + c) * 27 + t]);
}

// ---------------- Main conv: implicit GEMM, bf16 WMMA 16x16x32 -----------------------
// Block = 8 waves: 2 M-tiles x 4 N-groups (full 256 cout, 32 output positions).
// Per tap: async-DMA the 64 KB B-tile into LDS (double-buffered), overlap with WMMAs.
__global__ __launch_bounds__(256) void conv_wmma(const __bf16* __restrict__ xf,
                                                 const __bf16* __restrict__ wt,
                                                 const float*  __restrict__ bias,
                                                 float* __restrict__ out) {
  __shared__ __bf16 bsm[2 * TAPELEMS];           // 2 x 64 KB (WGP has 320 KB LDS)

  int tid  = threadIdx.x;
  int lane = tid & 31;
  int wid  = blockIdx.x * 8 + (tid >> 5);
  int mtile = wid >> 2;                          // 1728 tiles of 16 positions
  int ngrp  = wid & 3;                           // 4 groups of 64 cout
  int rrow  = lane & 15;
  int hi    = lane >> 4;
  int klo   = hi * 8;                            // ISA 16-bit A lane K-offset

  // this lane's M row -> input base coordinate (stride-2)
  int m  = mtile * 16 + rrow;
  int b  = m / SPATIAL;
  int s  = m % SPATIAL;
  int ow = s % OW;
  int oh = (s / OW) % OW;
  int od = s / (OW * OW);
  int ibase = ((b * FW + 2 * od) * FW + 2 * oh) * FW + 2 * ow;
  const __bf16* pA0 = xf + (size_t)ibase * 128 + klo;

  // async-copy bookkeeping: each thread moves 256 B (16 x b128) per tap
  uint32_t lds_base  = (uint32_t)(uintptr_t)(void*)bsm;
  uint32_t my_lds    = lds_base + (uint32_t)tid * 256u;
  uint32_t my_goff0  = (uint32_t)tid * 256u;     // byte offset within a tap tile
  // LDS base offset (bytes) of this wave's fragment row for colgroup (ngrp*4): per c0g/ns added later
  const __bf16* lfrag0 = bsm + ((size_t)(ngrp * 4) * 32 + lane) * 16;

  v8f acc[4] = {};

  // prologue: kick off tap 0 copy
  {
    uint32_t go = my_goff0;                      // + t*65536 with t=0
    uint32_t lo = my_lds;                        // buffer 0
    #pragma unroll
    for (int j = 0; j < 16; ++j) {
      asm volatile("global_load_async_to_lds_b128 %0, %1, %2"
                   :: "v"(lo + j * 16u), "v"(go + j * 16u), "s"(wt) : "memory");
    }
  }

  for (int t = 0; t < 27; ++t) {
    int kd = t / 9, kh = (t / 3) % 3, kw = t % 3;
    int dtap = (kd * FW + kh) * FW + kw;
    const __bf16* pAt = pA0 + (size_t)dtap * 128;

    asm volatile("s_wait_asynccnt 0x0" ::: "memory");  // my copies for tap t done
    __syncthreads();                                   // everyone's copies done

    if (t < 26) {                                      // kick off tap t+1 (other buffer)
      uint32_t go = my_goff0 + (uint32_t)(t + 1) * 65536u;
      uint32_t lo = my_lds + (uint32_t)((t + 1) & 1) * 65536u;
      #pragma unroll
      for (int j = 0; j < 16; ++j) {
        asm volatile("global_load_async_to_lds_b128 %0, %1, %2"
                     :: "v"(lo + j * 16u), "v"(go + j * 16u), "s"(wt) : "memory");
      }
    }

    const __bf16* lbuf = lfrag0 + (size_t)(t & 1) * TAPELEMS;
    #pragma unroll
    for (int c0g = 0; c0g < 4; ++c0g) {
      // A fragment: lane holds row m, K = klo..klo+7 and klo+16..klo+23 (global, L2-resident)
      const __bf16* pa = pAt + c0g * 32;
      v8bf alo = *(const v8bf*)(pa);
      v8bf ahi = *(const v8bf*)(pa + 16);
      v16bf a = __builtin_shufflevector(alo, ahi, 0,1,2,3,4,5,6,7,8,9,10,11,12,13,14,15);
      // B fragments from LDS: [c0g][colgroup][lane][16] -> contiguous 32 B per lane
      const __bf16* lb = lbuf + (size_t)c0g * (16 * 32 * 16);
      #pragma unroll
      for (int ns = 0; ns < 4; ++ns) {
        const __bf16* fp = lb + (size_t)ns * (32 * 16);
        v8bf blo = *(const v8bf*)(fp);
        v8bf bhi = *(const v8bf*)(fp + 8);
        v16bf bm = __builtin_shufflevector(blo, bhi, 0,1,2,3,4,5,6,7,8,9,10,11,12,13,14,15);
        acc[ns] = __builtin_amdgcn_wmma_f32_16x16x32_bf16(
            false, a, false, bm, (short)0, acc[ns], false, false);
      }
    }
    __syncthreads();   // all waves done reading buffer (t&1) before it is refilled at t+2
  }

  // Epilogue: lane's 8 acc VGPRs are rows (hi*8..hi*8+7), column (lane&15);
  // ow is the fastest M dim so the 8 rows are 8 contiguous floats -> two float4 stores.
  int mrow0 = mtile * 16 + hi * 8;
  int b2 = mrow0 / SPATIAL;
  int s0 = mrow0 % SPATIAL;
  #pragma unroll
  for (int ns = 0; ns < 4; ++ns) {
    int n = ngrp * 64 + ns * 16 + rrow;
    float bv = bias[n];
    size_t o0 = ((size_t)b2 * COUT + n) * SPATIAL + s0;
    f4 lo4 = {acc[ns][0] + bv, acc[ns][1] + bv, acc[ns][2] + bv, acc[ns][3] + bv};
    f4 hi4 = {acc[ns][4] + bv, acc[ns][5] + bv, acc[ns][6] + bv, acc[ns][7] + bv};
    *(f4*)(out + o0)     = lo4;
    *(f4*)(out + o0 + 4) = hi4;
  }
}

extern "C" void kernel_launch(void* const* d_in, const int* in_sizes, int n_in,
                              void* d_out, int out_size, void* d_ws, size_t ws_size,
                              hipStream_t stream) {
  (void)in_sizes; (void)n_in; (void)out_size; (void)ws_size;
  const float* x    = (const float*)d_in[0];
  const float* w    = (const float*)d_in[1];
  const float* bias = (const float*)d_in[2];
  float* out = (float*)d_out;

  char* ws = (char*)d_ws;
  // t1: (2,48,48,49,128) f32 = 115,605,504 B
  // t2: (2,48,49,49,128) f32 = 118,013,952 B
  // xf: (2,49,49,49,128) bf16 = 60,236,288 B  (reuses t1 region; t1 dead after pass 2)
  // wt: 27*256*128 bf16 = 1,769,472 B
  float*  t1 = (float*)ws;
  float*  t2 = (float*)(ws + 115605504ull);
  __bf16* xf = (__bf16*)ws;
  __bf16* wt = (__bf16*)(ws + 115605504ull + 118013952ull);

  fir_w_tr  <<<dim3(2 * 48 * 48 * 4), dim3(256), 0, stream>>>(x, t1);
  fir_h     <<<dim3(115248),          dim3(256), 0, stream>>>(t1, t2);
  fir_d_bf16<<<dim3(117649),          dim3(256), 0, stream>>>(t2, xf);
  wprep     <<<dim3(3456),            dim3(256), 0, stream>>>(w, wt);
  // 1728 M-tiles * 4 N-groups = 6912 waves / 8 waves per block
  conv_wmma <<<dim3(864),             dim3(256), 0, stream>>>(xf, wt, bias, out);
}
